// Predictor2D_56358560858590
// MI455X (gfx1250) — compile-verified, matrix operand order checked by reading, not run on hardware
//
#include <hip/hip_runtime.h>

// ---------------- problem constants ----------------
constexpr int cR = 128, cL = 384, cD = 128, cH = 8, cDH = 64, cDP = 128;
constexpr int cINNER = 512;            // H*DH
constexpr int cDHW = 16;               // D/H
constexpr int RL = cR * cL;            // 49152

typedef __attribute__((ext_vector_type(16))) __bf16 v16bf;
typedef __attribute__((ext_vector_type(8)))  float  v8f;

// ---------------- helpers ----------------
__device__ __forceinline__ unsigned short f32_to_bf16(float f) {
    unsigned int u = __float_as_uint(f);
    u = (u + 0x7FFFu + ((u >> 16) & 1u)) >> 16;
    return (unsigned short)u;
}
__device__ __forceinline__ float bf16_to_f32(unsigned short s) {
    return __uint_as_float(((unsigned int)s) << 16);
}
__device__ __forceinline__ v8f vzero8() {
    v8f z = {0.f, 0.f, 0.f, 0.f, 0.f, 0.f, 0.f, 0.f};
    return z;
}
__device__ __forceinline__ v8f wmma_bf16(v16bf a, v16bf b, v8f c) {
    // D = A(16x32 bf16) x B(32x16 bf16) + C(16x16 f32)
    return __builtin_amdgcn_wmma_f32_16x16x32_bf16(false, a, false, b,
                                                   (short)0, c, false, false);
}

// A-style fragment from row-major storage: lanes index rows (or, for B=X^T,
// lanes index rows of X). Per ISA: lane<16 -> K {0..7,16..23}; lane>=16 ->
// K {8..15,24..31}. Elements K-contiguous -> two 16B loads per lane.
__device__ __forceinline__ v16bf frag_rm(const unsigned short* p0, size_t stride, int lane) {
    union { v16bf v; uint4 q[2]; } u;
    const unsigned short* p = p0 + (size_t)(lane & 15) * stride + ((lane < 16) ? 0 : 8);
    u.q[0] = *reinterpret_cast<const uint4*>(p);
    u.q[1] = *reinterpret_cast<const uint4*>(p + 16);
    return u.v;
}
// B fragment from row-major B[k][n] (lanes index columns n, elements stride along k)
__device__ __forceinline__ v16bf frag_kn(const unsigned short* p0, size_t stride, int lane) {
    union { v16bf v; unsigned short s[16]; } u;
    const unsigned short* p = p0 + (lane & 15);
    int kA = (lane < 16) ? 0 : 8;
#pragma unroll
    for (int j = 0; j < 8; ++j) {
        u.s[j]     = p[(size_t)(kA + j) * stride];
        u.s[8 + j] = p[(size_t)(kA + 16 + j) * stride];
    }
    return u.v;
}

// ---------------- Tensor Data Mover (gfx1250) ----------------
#if defined(__gfx1250__) && __has_builtin(__builtin_amdgcn_tensor_load_to_lds)
#define USE_TDM 1
typedef unsigned int u32x4 __attribute__((ext_vector_type(4)));
typedef int i32x4 __attribute__((ext_vector_type(4)));
typedef int i32x8 __attribute__((ext_vector_type(8)));

// 2D tile load: tile_d0 contiguous elements (2B each) x tile_d1 rows, global
// row stride row_stride_elems, into LDS with 16B of pad appended per 128B row
// (-> 72-ushort LDS row pitch, matching frag_rm's bank-friendly layout).
// D# layout per CDNA5 ISA ch.8 (group0 128b, group1 256b; groups 2/3 unused for 2D).
__device__ __forceinline__ void tdm_load_2d_pad(const void* gsrc, unsigned lds_off,
                                                unsigned tile_d0, unsigned tile_d1,
                                                unsigned long long row_stride_elems) {
    unsigned long long ga = (unsigned long long)gsrc;
    u32x4 g0;
    g0[0] = 1u;                                                 // count=1, user desc
    g0[1] = lds_off;                                            // lds_addr (bytes)
    g0[2] = (unsigned)(ga & 0xFFFFFFFFull);                     // global_addr[31:0]
    g0[3] = (unsigned)((ga >> 32) & 0x1FFFFFFull) | (2u << 30); // addr[56:32] | type=2
    i32x8 g1;
    // data_size=2B(code1) | pad_enable | pad_interval=32dw(code4) | pad_amount=4dw(code3)
    g1[0] = (int)((1u << 16) | (1u << 20) | (4u << 22) | (3u << 25));
    const unsigned td0 = 0x7FFFFFFFu, td1 = 0x7FFFFFFFu;        // huge dims: no OOB clip
    g1[1] = (int)((td0 & 0xFFFFu) << 16);                       // tensor_dim0[15:0]
    g1[2] = (int)((td0 >> 16) | ((td1 & 0xFFFFu) << 16));       // td0[31:16] | td1[15:0]
    g1[3] = (int)((td1 >> 16) | (tile_d0 << 16));               // td1[31:16] | tile_dim0
    g1[4] = (int)(tile_d1 & 0xFFFFu);                           // tile_dim1 (tile_dim2=0)
    g1[5] = (int)(row_stride_elems & 0xFFFFFFFFull);            // tensor_dim0_stride lo
    g1[6] = (int)((row_stride_elems >> 32) & 0xFFFFull);        // stride hi | td1_stride=0
    g1[7] = 0;
    i32x4 z4 = {0, 0, 0, 0};
#if defined(__clang_major__) && __clang_major__ >= 23
    i32x8 z8 = {0, 0, 0, 0, 0, 0, 0, 0};
    __builtin_amdgcn_tensor_load_to_lds(g0, g1, z4, z4, z8, 0);
#else
    __builtin_amdgcn_tensor_load_to_lds(g0, g1, z4, z4, 0);
#endif
}
#endif

// ---------------- f32 -> bf16 conversion ----------------
__global__ void cvt_bf16_kernel(const float* __restrict__ src,
                                unsigned short* __restrict__ dst, int n) {
    int i = blockIdx.x * 256 + threadIdx.x;
    if (i < n) dst[i] = f32_to_bf16(src[i]);
}

// ---------------- generic WMMA GEMM: C(MxN) = A(MxK) @ B(KxN) ----------------
// bf16 A,B row-major; M%128==0 (grid.y), N%128==0 (grid.x), K%32==0.
template <bool OUT_BF16>
__global__ __launch_bounds__(256) void gemm_kernel(const unsigned short* __restrict__ A,
                                                   const unsigned short* __restrict__ B,
                                                   void* __restrict__ C,
                                                   int K, int lda, int ldb, int ldc) {
    __shared__ __align__(16) unsigned short As[128 * 40];
    __shared__ __align__(16) unsigned short Bs[32 * 136];
    const int tid = threadIdx.x, lane = tid & 31, wid = tid >> 5;
    const int wm = (wid & 3) * 32, wn = (wid >> 2) * 64;
    const int mBase = blockIdx.y * 128, nBase = blockIdx.x * 128;

    v8f acc[2][4];
#pragma unroll
    for (int mt = 0; mt < 2; ++mt)
#pragma unroll
        for (int nt = 0; nt < 4; ++nt) acc[mt][nt] = vzero8();

    for (int k0 = 0; k0 < K; k0 += 32) {
        __syncthreads();
        for (int idx = tid; idx < 128 * 8; idx += 256) {
            int row = idx >> 3, c4 = (idx & 7) * 4;
            *reinterpret_cast<uint2*>(&As[row * 40 + c4]) =
                *reinterpret_cast<const uint2*>(&A[(size_t)(mBase + row) * lda + k0 + c4]);
        }
        for (int idx = tid; idx < 32 * 32; idx += 256) {
            int row = idx >> 5, c4 = (idx & 31) * 4;
            *reinterpret_cast<uint2*>(&Bs[row * 136 + c4]) =
                *reinterpret_cast<const uint2*>(&B[(size_t)(k0 + row) * ldb + nBase + c4]);
        }
        if (k0 + 32 < K) {   // prefetch next stage into cache (global_prefetch_b8)
            __builtin_prefetch(&A[(size_t)(mBase + (tid >> 1)) * lda + k0 + 32 + (tid & 1) * 16], 0, 0);
            __builtin_prefetch(&B[(size_t)(k0 + 32 + (tid >> 3)) * ldb + nBase + (tid & 7) * 16], 0, 0);
        }
        __syncthreads();
        v16bf a[2], b[4];
#pragma unroll
        for (int mt = 0; mt < 2; ++mt) a[mt] = frag_rm(As + (wm + mt * 16) * 40, 40, lane);
#pragma unroll
        for (int nt = 0; nt < 4; ++nt) b[nt] = frag_kn(Bs + wn + nt * 16, 136, lane);
#pragma unroll
        for (int mt = 0; mt < 2; ++mt)
#pragma unroll
            for (int nt = 0; nt < 4; ++nt) acc[mt][nt] = wmma_bf16(a[mt], b[nt], acc[mt][nt]);
    }
    const int rsel = (lane < 16) ? 0 : 8, cc = lane & 15;
#pragma unroll
    for (int mt = 0; mt < 2; ++mt)
#pragma unroll
        for (int nt = 0; nt < 4; ++nt)
#pragma unroll
            for (int i = 0; i < 8; ++i) {
                size_t row = mBase + wm + mt * 16 + rsel + i;
                size_t col = nBase + wn + nt * 16 + cc;
                if (OUT_BF16)
                    ((unsigned short*)C)[row * ldc + col] = f32_to_bf16(acc[mt][nt][i]);
                else
                    ((float*)C)[row * ldc + col] = acc[mt][nt][i];
            }
}

// ---------------- column attention: one block per (l,h) ----------------
// qw: (RL x INNER) bf16 at row r*L+l; kvw: (RL x 2*INNER) K|V. Output bf16.
__global__ __launch_bounds__(256) void col_attn_kernel(const unsigned short* __restrict__ qw,
                                                       const unsigned short* __restrict__ kvw,
                                                       unsigned short* __restrict__ ocol) {
    __shared__ __align__(16) float S[128 * 128];
    __shared__ __align__(16) unsigned short P[128 * 128];
    __shared__ __align__(16) unsigned short Vs[128 * 72];
    const int l = blockIdx.x % cL, h = blockIdx.x / cL;
    const int tid = threadIdx.x, lane = tid & 31, wid = tid >> 5;
    const int rsel = (lane < 16) ? 0 : 8, cc = lane & 15;
    const size_t qstr = (size_t)cL * cINNER;       // row stride over r in q buffer
    const size_t kstr = (size_t)cL * 2 * cINNER;   // row stride over r in kv buffer
    const unsigned short* qbase = qw + (size_t)l * cINNER + h * cDH;
    const unsigned short* kbase = kvw + (size_t)l * 2 * cINNER + h * cDH;
    const unsigned short* vbase = kbase + cINNER;

    // stage V (128 x 64) into LDS once (coalesced); consumed after two barriers
    for (int idx = tid; idx < 128 * 16; idx += 256) {
        int row = idx >> 4, c4 = (idx & 15) * 4;
        *reinterpret_cast<uint2*>(&Vs[row * 72 + c4]) =
            *reinterpret_cast<const uint2*>(&vbase[(size_t)row * kstr + c4]);
    }

    // S = Q K^T * scale : wave w owns 16-row strip; Q fragments hoisted
    v16bf qa[2];
#pragma unroll
    for (int ks = 0; ks < 2; ++ks)
        qa[ks] = frag_rm(qbase + (size_t)(wid * 16) * qstr + ks * 32, qstr, lane);
#pragma unroll 2
    for (int ct = 0; ct < 8; ++ct) {
        v8f acc = vzero8();
#pragma unroll
        for (int ks = 0; ks < 2; ++ks) {
            v16bf b = frag_rm(kbase + (size_t)(ct * 16) * kstr + ks * 32, kstr, lane); // K^T
            acc = wmma_bf16(qa[ks], b, acc);
        }
#pragma unroll
        for (int i = 0; i < 8; ++i)
            S[(wid * 16 + rsel + i) * 128 + ct * 16 + cc] = acc[i] * 0.125f;
    }
    __syncthreads();
    // per-row softmax (128 rows, threads 0..127)
    if (tid < 128) {
        float* row = &S[tid * 128];
        float mx = -1e30f;
        for (int j = 0; j < 128; ++j) mx = fmaxf(mx, row[j]);
        float sm = 0.f;
        for (int j = 0; j < 128; ++j) { float e = __expf(row[j] - mx); sm += e; row[j] = e; }
        float inv = 1.f / sm;
        for (int j = 0; j < 128; ++j) P[tid * 128 + j] = f32_to_bf16(row[j] * inv);
    }
    __syncthreads();
    // O = P V  (128x64); P fragments hoisted across output-column tiles
    v16bf pa[4];
#pragma unroll
    for (int ks = 0; ks < 4; ++ks)
        pa[ks] = frag_rm(P + (size_t)(wid * 16) * 128 + ks * 32, 128, lane);
#pragma unroll
    for (int nt = 0; nt < 4; ++nt) {
        v8f acc = vzero8();
#pragma unroll
        for (int ks = 0; ks < 4; ++ks) {
            v16bf b = frag_kn(Vs + (size_t)(ks * 32) * 72 + nt * 16, 72, lane);
            acc = wmma_bf16(pa[ks], b, acc);
        }
#pragma unroll
        for (int i = 0; i < 8; ++i) {
            int r = wid * 16 + rsel + i;
            ocol[((size_t)r * cL + l) * cINNER + h * cDH + nt * 16 + cc] = f32_to_bf16(acc[i]);
        }
    }
}

// ---------------- pair bias: LN(pair[i,j,:]) @ Wpair -> pb[i,j,h] ----------------
__global__ __launch_bounds__(256) void pair_bias_kernel(const float* __restrict__ pair,
                                                        const float* __restrict__ ln_g,
                                                        const float* __restrict__ ln_b,
                                                        const float* __restrict__ Wpair,
                                                        float* __restrict__ pb) {
    const int wid = threadIdx.x >> 5, lane = threadIdx.x & 31;
    const int p = blockIdx.x * 8 + wid;   // (i*L + j)
    const float* x = pair + (size_t)p * cDP;
    float v[4];
#pragma unroll
    for (int e = 0; e < 4; ++e) v[e] = x[lane * 4 + e];
    float s = v[0] + v[1] + v[2] + v[3];
#pragma unroll
    for (int m = 16; m >= 1; m >>= 1) s += __shfl_xor(s, m, 32);
    float mu = s * (1.f / 128.f);
    float vs = 0.f;
#pragma unroll
    for (int e = 0; e < 4; ++e) { float d = v[e] - mu; vs += d * d; }
#pragma unroll
    for (int m = 16; m >= 1; m >>= 1) vs += __shfl_xor(vs, m, 32);
    float rstd = rsqrtf(vs * (1.f / 128.f) + 1e-5f);
    float nrm[4];
#pragma unroll
    for (int e = 0; e < 4; ++e) {
        int c = lane * 4 + e;
        nrm[e] = (v[e] - mu) * rstd * ln_g[c] + ln_b[c];
    }
#pragma unroll
    for (int h = 0; h < cH; ++h) {
        float d = 0.f;
#pragma unroll
        for (int e = 0; e < 4; ++e) d += nrm[e] * Wpair[(lane * 4 + e) * cH + h];
#pragma unroll
        for (int m = 16; m >= 1; m >>= 1) d += __shfl_xor(d, m, 32);
        if (lane == 0) pb[(size_t)p * cH + h] = d;
    }
}

// ---------------- tie weights: softmax over r of (qs+bsq).(ks+bsk)/sqrt(16) ----------------
__global__ __launch_bounds__(128) void tie_kernel(const float* __restrict__ qs,
                                                  const float* __restrict__ ksb,
                                                  const float* __restrict__ bsq,
                                                  const float* __restrict__ bsk,
                                                  float* __restrict__ wtie) {
    __shared__ float red[128];
    const int l = blockIdx.x % cL, h = blockIdx.x / cL;
    const int r = threadIdx.x;
    float dot = 0.f;
#pragma unroll
    for (int d = 0; d < cDHW; ++d) {
        int c = h * cDHW + d;
        dot += (qs[(size_t)l * cD + c] + bsq[c]) *
               (ksb[((size_t)r * cL + l) * cD + c] + bsk[c]);
    }
    float logit = dot * 0.25f;   // DHW^-0.5
    red[r] = logit; __syncthreads();
    for (int st = 64; st >= 1; st >>= 1) {
        if (r < st) red[r] = fmaxf(red[r], red[r + st]);
        __syncthreads();
    }
    float mx = red[0]; __syncthreads();
    float e = __expf(logit - mx);
    red[r] = e; __syncthreads();
    for (int st = 64; st >= 1; st >>= 1) {
        if (r < st) red[r] += red[r + st];
        __syncthreads();
    }
    wtie[((size_t)l * cH + h) * cR + r] = e / red[0];
}

// ---------------- qt = wtie * qh (in place on bf16 q_h buffer) ----------------
__global__ void qt_scale_kernel(unsigned short* __restrict__ qh,
                                const float* __restrict__ wtie) {
    size_t idx = (size_t)blockIdx.x * 256 + threadIdx.x;
    if (idx >= (size_t)RL * cINNER) return;
    int r = (int)(idx / ((size_t)cL * cINNER));
    int l = (int)((idx / cINNER) % cL);
    int c = (int)(idx % cINNER);
    int h = c >> 6;
    float v = bf16_to_f32(qh[idx]) * wtie[((size_t)l * cH + h) * cR + r];
    qh[idx] = f32_to_bf16(v);
}

// ---------------- dots[h,i,j] = sum_r sum_d qt[r,h,i,d]*kh[r,h,j,d]  * scale ----------------
__global__ __launch_bounds__(256) void dots_kernel(const unsigned short* __restrict__ qt,
                                                   const unsigned short* __restrict__ kvh,
                                                   float* __restrict__ dots) {
    __shared__ __align__(16) unsigned short Aq[128 * 72];
    __shared__ __align__(16) unsigned short Bk[128 * 72];
    const int jb = blockIdx.x, ib = blockIdx.y, h = blockIdx.z;
    const int tid = threadIdx.x, lane = tid & 31, wid = tid >> 5;
    const int wm = (wid & 3) * 32, wn = (wid >> 2) * 64;
    v8f acc[2][4];
#pragma unroll
    for (int mt = 0; mt < 2; ++mt)
#pragma unroll
        for (int nt = 0; nt < 4; ++nt) acc[mt][nt] = vzero8();

    for (int r = 0; r < cR; ++r) {
        __syncthreads();
#ifdef USE_TDM
        // Tensor Data Mover: 64x128 2B tiles (row strides INNER / 2*INNER) into
        // padded LDS (72-ushort pitch via TDM pad), one issuing wave.
        if (wid == 0) {
            tdm_load_2d_pad(&qt[((size_t)r * cL + ib * 128) * cINNER + h * cDH],
                            (unsigned)(unsigned long long)(void*)Aq, 64, 128, cINNER);
            tdm_load_2d_pad(&kvh[((size_t)r * cL + jb * 128) * (2 * cINNER) + h * cDH],
                            (unsigned)(unsigned long long)(void*)Bk, 64, 128, 2 * cINNER);
            __builtin_amdgcn_s_wait_tensorcnt(0);
        }
#else
        for (int idx = tid; idx < 128 * 16; idx += 256) {
            int row = idx >> 4, c4 = (idx & 15) * 4;
            *reinterpret_cast<uint2*>(&Aq[row * 72 + c4]) =
                *reinterpret_cast<const uint2*>(
                    &qt[((size_t)r * cL + ib * 128 + row) * cINNER + h * cDH + c4]);
            *reinterpret_cast<uint2*>(&Bk[row * 72 + c4]) =
                *reinterpret_cast<const uint2*>(
                    &kvh[((size_t)r * cL + jb * 128 + row) * (2 * cINNER) + h * cDH + c4]);
        }
#endif
        __syncthreads();
#pragma unroll
        for (int ks = 0; ks < cDH; ks += 32) {
            v16bf a[2], b[4];
#pragma unroll
            for (int mt = 0; mt < 2; ++mt) a[mt] = frag_rm(Aq + (wm + mt * 16) * 72 + ks, 72, lane);
#pragma unroll
            for (int nt = 0; nt < 4; ++nt) b[nt] = frag_rm(Bk + (wn + nt * 16) * 72 + ks, 72, lane);
#pragma unroll
            for (int mt = 0; mt < 2; ++mt)
#pragma unroll
                for (int nt = 0; nt < 4; ++nt) acc[mt][nt] = wmma_bf16(a[mt], b[nt], acc[mt][nt]);
        }
    }
    const int rsel = (lane < 16) ? 0 : 8, cc = lane & 15;
#pragma unroll
    for (int mt = 0; mt < 2; ++mt)
#pragma unroll
        for (int nt = 0; nt < 4; ++nt)
#pragma unroll
            for (int i = 0; i < 8; ++i)
                dots[((size_t)h * cL + ib * 128 + wm + mt * 16 + rsel + i) * cL +
                     jb * 128 + wn + nt * 16 + cc] = acc[mt][nt][i] * 0.125f;
}

// ---------------- attn_h = softmax_j(dots + pb) -> bf16 ----------------
__global__ __launch_bounds__(128) void attn_softmax_kernel(const float* __restrict__ dots,
                                                           const float* __restrict__ pb,
                                                           unsigned short* __restrict__ attnh) {
    __shared__ float red[128];
    const int i = blockIdx.x % cL, h = blockIdx.x / cL;
    const int t = threadIdx.x;
    float lv[3];
    float mx = -1e30f;
#pragma unroll
    for (int u = 0; u < 3; ++u) {
        int j = t + u * 128;
        lv[u] = dots[((size_t)h * cL + i) * cL + j] + pb[((size_t)i * cL + j) * cH + h];
        mx = fmaxf(mx, lv[u]);
    }
    red[t] = mx; __syncthreads();
    for (int st = 64; st >= 1; st >>= 1) {
        if (t < st) red[t] = fmaxf(red[t], red[t + st]);
        __syncthreads();
    }
    mx = red[0]; __syncthreads();
    float sm = 0.f, ev[3];
#pragma unroll
    for (int u = 0; u < 3; ++u) { ev[u] = __expf(lv[u] - mx); sm += ev[u]; }
    red[t] = sm; __syncthreads();
    for (int st = 64; st >= 1; st >>= 1) {
        if (t < st) red[t] += red[t + st];
        __syncthreads();
    }
    float inv = 1.f / red[0];
#pragma unroll
    for (int u = 0; u < 3; ++u)
        attnh[((size_t)h * cL + i) * cL + t + u * 128] = f32_to_bf16(ev[u] * inv);
}

// ---------------- oh[r,h] = attn_h[h] @ V[r,h]  (128-row tile per block) ----------------
__global__ __launch_bounds__(256) void row_av_kernel(const unsigned short* __restrict__ attnh,
                                                     const unsigned short* __restrict__ kvh,
                                                     unsigned short* __restrict__ orow) {
    __shared__ __align__(16) unsigned short Pa[128 * 40];
    __shared__ __align__(16) unsigned short Vb[32 * 72];
    const int ib = blockIdx.x;           // 0..2  (i tile)
    const int r = blockIdx.y / cH, h = blockIdx.y % cH;
    const int tid = threadIdx.x, lane = tid & 31, wid = tid >> 5;
    const unsigned short* abase = attnh + ((size_t)h * cL + ib * 128) * cL;
    const unsigned short* vbase = kvh + ((size_t)r * cL) * (2 * cINNER) + cINNER + h * cDH;
    v8f acc[4];
#pragma unroll
    for (int nt = 0; nt < 4; ++nt) acc[nt] = vzero8();

    for (int ks = 0; ks < cL; ks += 32) {
        __syncthreads();
        for (int idx = tid; idx < 128 * 8; idx += 256) {
            int row = idx >> 3, c4 = (idx & 7) * 4;
            *reinterpret_cast<uint2*>(&Pa[row * 40 + c4]) =
                *reinterpret_cast<const uint2*>(&abase[(size_t)row * cL + ks + c4]);
        }
        for (int idx = tid; idx < 32 * 16; idx += 256) {
            int row = idx >> 4, c4 = (idx & 15) * 4;
            *reinterpret_cast<uint2*>(&Vb[row * 72 + c4]) =
                *reinterpret_cast<const uint2*>(&vbase[(size_t)(ks + row) * (2 * cINNER) + c4]);
        }
        __syncthreads();
        v16bf a = frag_rm(Pa + (size_t)(wid * 16) * 40, 40, lane);
#pragma unroll
        for (int nt = 0; nt < 4; ++nt) {
            v16bf b = frag_kn(Vb + nt * 16, 72, lane);
            acc[nt] = wmma_bf16(a, b, acc[nt]);
        }
    }
    const int rsel = (lane < 16) ? 0 : 8, cc = lane & 15;
#pragma unroll
    for (int nt = 0; nt < 4; ++nt)
#pragma unroll
        for (int i = 0; i < 8; ++i)
            orow[((size_t)r * cL + ib * 128 + wid * 16 + rsel + i) * cINNER +
                 h * cDH + nt * 16 + cc] = f32_to_bf16(acc[nt][i]);
}

// ---------------- final combine: 0.5*((ocol@Wo_w + bo_w)+(orow@Wo_h + bo_h)) ----------------
__global__ void combine_kernel(const float* __restrict__ outw, const float* __restrict__ outh,
                               const float* __restrict__ bo_w, const float* __restrict__ bo_h,
                               float* __restrict__ out) {
    size_t idx = (size_t)blockIdx.x * 256 + threadIdx.x;
    if (idx >= (size_t)RL * cD) return;
    int c = (int)(idx % cD);
    out[idx] = 0.5f * (outw[idx] + bo_w[c] + outh[idx] + bo_h[c]);
}

// ---------------- host launcher ----------------
extern "C" void kernel_launch(void* const* d_in, const int* in_sizes, int n_in,
                              void* d_out, int out_size, void* d_ws, size_t ws_size,
                              hipStream_t stream) {
    const float* m     = (const float*)d_in[0];
    const float* pair  = (const float*)d_in[1];
    const float* Wq_w  = (const float*)d_in[2];
    const float* Wkv_w = (const float*)d_in[3];
    const float* Wo_w  = (const float*)d_in[4];
    const float* bo_w  = (const float*)d_in[5];
    const float* Wq_h  = (const float*)d_in[6];
    const float* Wkv_h = (const float*)d_in[7];
    const float* Wo_h  = (const float*)d_in[8];
    const float* bo_h  = (const float*)d_in[9];
    const float* ln_g  = (const float*)d_in[10];
    const float* ln_b  = (const float*)d_in[11];
    const float* Wpair = (const float*)d_in[12];
    const float* Wsq   = (const float*)d_in[13];
    const float* bsq   = (const float*)d_in[14];
    const float* Wsk   = (const float*)d_in[15];
    const float* bsk   = (const float*)d_in[16];
    float* out = (float*)d_out;

    char* ws = (char*)d_ws;
    size_t off = 0;
    auto alloc = [&](size_t elems, size_t esz) -> void* {
        off = (off + 255) & ~(size_t)255;
        void* p = ws + off;
        off += elems * esz;
        return p;
    };
    unsigned short* mb    = (unsigned short*)alloc((size_t)RL * cD, 2);
    unsigned short* wqw_b = (unsigned short*)alloc((size_t)cD * cINNER, 2);
    unsigned short* wkvw_b= (unsigned short*)alloc((size_t)cD * 2 * cINNER, 2);
    unsigned short* wow_b = (unsigned short*)alloc((size_t)cINNER * cD, 2);
    unsigned short* wqh_b = (unsigned short*)alloc((size_t)cD * cINNER, 2);
    unsigned short* wkvh_b= (unsigned short*)alloc((size_t)cD * 2 * cINNER, 2);
    unsigned short* woh_b = (unsigned short*)alloc((size_t)cINNER * cD, 2);
    unsigned short* wsq_b = (unsigned short*)alloc((size_t)cD * cD, 2);
    unsigned short* wsk_b = (unsigned short*)alloc((size_t)cD * cD, 2);
    unsigned short* qw    = (unsigned short*)alloc((size_t)RL * cINNER, 2);
    unsigned short* kvw   = (unsigned short*)alloc((size_t)RL * 2 * cINNER, 2);
    unsigned short* qh    = (unsigned short*)alloc((size_t)RL * cINNER, 2);
    unsigned short* kvh   = (unsigned short*)alloc((size_t)RL * 2 * cINNER, 2);
    float* qs   = (float*)alloc((size_t)cL * cD, 4);
    float* ksb  = (float*)alloc((size_t)RL * cD, 4);
    float* wtie = (float*)alloc((size_t)cL * cH * cR, 4);
    float* pb   = (float*)alloc((size_t)cL * cL * cH, 4);
    float* dots = (float*)alloc((size_t)cH * cL * cL, 4);
    unsigned short* attnh = (unsigned short*)alloc((size_t)cH * cL * cL, 2);
    unsigned short* ocol  = (unsigned short*)alloc((size_t)RL * cINNER, 2);
    unsigned short* orow  = (unsigned short*)alloc((size_t)RL * cINNER, 2);
    float* outw = (float*)alloc((size_t)RL * cD, 4);
    float* outh = (float*)alloc((size_t)RL * cD, 4);
    (void)ws_size; (void)n_in; (void)in_sizes; (void)out_size;

    auto cvt = [&](const float* s, unsigned short* d, int n) {
        cvt_bf16_kernel<<<(n + 255) / 256, 256, 0, stream>>>(s, d, n);
    };
    // 1) bf16 conversions
    cvt(m, mb, RL * cD);
    cvt(Wq_w, wqw_b, cD * cINNER);
    cvt(Wkv_w, wkvw_b, cD * 2 * cINNER);
    cvt(Wo_w, wow_b, cINNER * cD);
    cvt(Wq_h, wqh_b, cD * cINNER);
    cvt(Wkv_h, wkvh_b, cD * 2 * cINNER);
    cvt(Wo_h, woh_b, cINNER * cD);
    cvt(Wsq, wsq_b, cD * cD);
    cvt(Wsk, wsk_b, cD * cD);

    // 2) projections (WMMA GEMMs, bf16 out)
    gemm_kernel<true><<<dim3(cINNER / 128, RL / 128), 256, 0, stream>>>(mb, wqw_b, qw, cD, cD, cINNER, cINNER);
    gemm_kernel<true><<<dim3(2 * cINNER / 128, RL / 128), 256, 0, stream>>>(mb, wkvw_b, kvw, cD, cD, 2 * cINNER, 2 * cINNER);
    gemm_kernel<true><<<dim3(cINNER / 128, RL / 128), 256, 0, stream>>>(mb, wqh_b, qh, cD, cD, cINNER, cINNER);
    gemm_kernel<true><<<dim3(2 * cINNER / 128, RL / 128), 256, 0, stream>>>(mb, wkvh_b, kvh, cD, cD, 2 * cINNER, 2 * cINNER);
    // tie-weight projections (f32 out): qs uses only row-0 of m (first L rows of mb)
    gemm_kernel<false><<<dim3(1, cL / 128), 256, 0, stream>>>(mb, wsq_b, qs, cD, cD, cD, cD);
    gemm_kernel<false><<<dim3(1, RL / 128), 256, 0, stream>>>(mb, wsk_b, ksb, cD, cD, cD, cD);

    // 3) pair bias + tie softmax
    pair_bias_kernel<<<(cL * cL) / 8, 256, 0, stream>>>(pair, ln_g, ln_b, Wpair, pb);
    tie_kernel<<<cL * cH, 128, 0, stream>>>(qs, ksb, bsq, bsk, wtie);

    // 4) column attention (WMMA QK^T, softmax, WMMA PV)
    col_attn_kernel<<<cL * cH, 256, 0, stream>>>(qw, kvw, ocol);

    // 5) row attention
    qt_scale_kernel<<<(unsigned)(((size_t)RL * cINNER + 255) / 256), 256, 0, stream>>>(qh, wtie);
    dots_kernel<<<dim3(cL / 128, cL / 128, cH), 256, 0, stream>>>(qh, kvh, dots);
    attn_softmax_kernel<<<cH * cL, 128, 0, stream>>>(dots, pb, attnh);
    row_av_kernel<<<dim3(cL / 128, cR * cH), 256, 0, stream>>>(attnh, kvh, orow);

    // 6) output projections + combine
    gemm_kernel<false><<<dim3(cD / 128, RL / 128), 256, 0, stream>>>(ocol, wow_b, outw, cINNER, cINNER, cD, cD);
    gemm_kernel<false><<<dim3(cD / 128, RL / 128), 256, 0, stream>>>(orow, woh_b, outh, cINNER, cINNER, cD, cD);
    combine_kernel<<<(unsigned)(((size_t)RL * cD + 255) / 256), 256, 0, stream>>>(outw, outh, bo_w, bo_h, out);
}
